// StepPredictor_66795331387603
// MI455X (gfx1250) — matrix operand order repaired and checked by reading, not compiled, not run on hardware
//
#include <hip/hip_runtime.h>
#include <hip/hip_bf16.h>

typedef __attribute__((ext_vector_type(16))) _Float16 v16h;
typedef __attribute__((ext_vector_type(8)))  _Float16 v8h;
typedef __attribute__((ext_vector_type(8)))  float    v8f;

#define MT 4   // 16-row tiles per wave (M-blocking)

// ---------------------------------------------------------------- utilities
__global__ void k_zero(float* __restrict__ p, int n) {
    int i = blockIdx.x * blockDim.x + threadIdx.x;
    if (i < n) p[i] = 0.f;
}

// Pack a row-major f32 weight W[K][Nout] into WMMA B-operand tiles.
// Tile (kt,nt) occupies 512 f16 at base (kt*Nt+nt)*512; lane l owns 16
// contiguous f16 = column n = nt*16 + (l&15), K = kt*32 + 16*(l>=16) + j.
__global__ void k_pack_w(const float* __restrict__ W, _Float16* __restrict__ dst,
                         int K, int Nout, int Kt, int Nt) {
    int idx = blockIdx.x * blockDim.x + threadIdx.x;
    int total = Kt * Nt * 512;
    if (idx >= total) return;
    int j    = idx & 15;
    int lane = (idx >> 4) & 31;
    int tile = idx >> 9;
    int nt = tile % Nt, kt = tile / Nt;
    int k = kt * 32 + ((lane >> 4) << 4) + j;
    int n = nt * 16 + (lane & 15);
    float v = (k < K && n < Nout) ? W[(size_t)k * Nout + n] : 0.f;
    dst[idx] = (_Float16)v;
}

// ---------------------------------------------------------------- graph prep
__global__ void k_deg(const int* __restrict__ dst, float* __restrict__ deg, int E) {
    int e = blockIdx.x * blockDim.x + threadIdx.x;
    if (e < E) atomicAdd(&deg[dst[e]], 1.f);
}

// deg -> dinv (in place) and selfnorm = dinv^2   (self-loop: deg+1)
__global__ void k_dinv(float* __restrict__ degdinv, float* __restrict__ selfn, int n) {
    int i = blockIdx.x * blockDim.x + threadIdx.x;
    if (i >= n) return;
    float d = rsqrtf(degdinv[i] + 1.f);
    degdinv[i] = d;
    selfn[i]   = d * d;
}

__global__ void k_norm(const int* __restrict__ src, const int* __restrict__ dst,
                       const float* __restrict__ dinv, float* __restrict__ norm, int E) {
    int e = blockIdx.x * blockDim.x + threadIdx.x;
    if (e < E) norm[e] = dinv[src[e]] * dinv[dst[e]];
}

// x [N,20] f32 -> act [N,32] f16 (zero padded K)
__global__ void k_cvt_x(const float* __restrict__ x, _Float16* __restrict__ act, int N) {
    int idx = blockIdx.x * blockDim.x + threadIdx.x;
    if (idx >= N * 32) return;
    int i = idx >> 5, k = idx & 31;
    act[idx] = (_Float16)(k < 20 ? x[(size_t)i * 20 + k] : 0.f);
}

// ---------------------------------------------------------------- WMMA GEMM
// C[M,Nout] = A[M, Kt*32] (f16 row-major, lda) x Bpacked (+bias, +relu).
// One wave computes MT consecutive 16x16 tiles for one nt: B loaded once per
// kt, reused by MT back-to-back WMMAs on independent accumulators (no D->A/B
// RAW hazards, loads of tile t+1 overlap WMMA of tile t).
__global__ void k_gemm(const _Float16* __restrict__ A, int lda,
                       const _Float16* __restrict__ Bp,
                       const float* __restrict__ bias,
                       float* __restrict__ Cf, int ldc,
                       _Float16* __restrict__ Ch, int ldh,
                       int M, int Kt, int Nt, int Nout, int relu) {
    int wave = (blockIdx.x * blockDim.x + threadIdx.x) >> 5;
    int lane = threadIdx.x & 31;
    int mtiles  = (M + 15) >> 4;
    int mgroups = (mtiles + MT - 1) / MT;
    if (wave >= mgroups * Nt) return;      // wave-uniform: EXEC stays all-ones
    int nt = wave % Nt;
    int mg = wave / Nt;
    int mbase = mg * (16 * MT);

    const _Float16* Ar[MT];
#pragma unroll
    for (int t = 0; t < MT; ++t) {
        int r = mbase + t * 16 + (lane & 15);
        if (r >= M) r = M - 1;             // clamp; masked at store
        Ar[t] = A + (size_t)r * lda + ((lane >> 4) << 4);
    }

    v8f c[MT];
#pragma unroll
    for (int t = 0; t < MT; ++t) c[t] = (v8f){};

    for (int kt = 0; kt < Kt; ++kt) {
        const _Float16* bp = Bp + ((size_t)(kt * Nt + nt) << 9) + (lane << 4);
        v8h b0 = *(const v8h*)(bp);
        v8h b1 = *(const v8h*)(bp + 8);
        v16h b;
#pragma unroll
        for (int j = 0; j < 8; ++j) { b[j] = b0[j]; b[8 + j] = b1[j]; }

        v16h a[MT];
#pragma unroll
        for (int t = 0; t < MT; ++t) {
            v8h a0 = *(const v8h*)(Ar[t] + kt * 32);
            v8h a1 = *(const v8h*)(Ar[t] + kt * 32 + 8);
#pragma unroll
            for (int j = 0; j < 8; ++j) { a[t][j] = a0[j]; a[t][8 + j] = a1[j]; }
        }
#pragma unroll
        for (int t = 0; t < MT; ++t)
            c[t] = __builtin_amdgcn_wmma_f32_16x16x32_f16(false, a[t], false, b,
                                                          (short)0, c[t], false, false);
    }

    int col = nt * 16 + (lane & 15);
    if (col < Nout) {
        float bv = bias ? bias[col] : 0.f;
#pragma unroll
        for (int t = 0; t < MT; ++t) {
            int rbase = mbase + t * 16 + ((lane >> 4) << 3);
#pragma unroll
            for (int r = 0; r < 8; ++r) {
                int row = rbase + r;
                if (row < M) {
                    float v = c[t][r] + bv;
                    if (relu) v = fmaxf(v, 0.f);
                    if (Cf) Cf[(size_t)row * ldc + col] = v;
                    if (Ch) Ch[(size_t)row * ldh + col] = (_Float16)v;
                }
            }
        }
    }
}

// ---------------------------------------------------------------- GCN aggregate
// agg[i,f] = h[i,f]*selfnorm[i] + bias[f]   (self-loop term + bias)
__global__ void k_agg_init(const float* __restrict__ h, const float* __restrict__ selfn,
                           const float* __restrict__ b, float* __restrict__ agg,
                           int N, int F) {
    int idx = blockIdx.x * blockDim.x + threadIdx.x;
    if (idx >= N * F) return;
    int i = idx / F, f = idx - i * F;
    agg[idx] = h[idx] * selfn[i] + b[f];
}

// agg[dst] += h[src]*norm   — one thread per (edge, float4 chunk)
__global__ void k_scatter(const int* __restrict__ src, const int* __restrict__ dst,
                          const float* __restrict__ norm, const float* __restrict__ h,
                          float* __restrict__ agg, int E, int lf4) {
    int idx = blockIdx.x * blockDim.x + threadIdx.x;
    if (idx >= (E << lf4)) return;
    int e  = idx >> lf4;
    int f4 = idx & ((1 << lf4) - 1);
    int F  = 4 << lf4;
    float w = norm[e];
    const float4 v = *(const float4*)(h + (size_t)src[e] * F + f4 * 4);
    float* o = agg + (size_t)dst[e] * F + f4 * 4;
    atomicAdd(o + 0, v.x * w);
    atomicAdd(o + 1, v.y * w);
    atomicAdd(o + 2, v.z * w);
    atomicAdd(o + 3, v.w * w);
}

// ---------------------------------------------------------------- batch norm
__global__ void k_bn_stats(const float* __restrict__ agg, float* __restrict__ sums,
                           float* __restrict__ sumsq, int N, int F) {
    int c   = threadIdx.x % F;
    int rg  = threadIdx.x / F;
    int rpb = blockDim.x / F;
    int row = blockIdx.x * rpb + rg;
    int stride = gridDim.x * rpb;
    float s = 0.f, q = 0.f;
    for (; row < N; row += stride) {
        float v = agg[(size_t)row * F + c];
        s += v; q += v * v;
    }
    atomicAdd(&sums[c], s);
    atomicAdd(&sumsq[c], q);
}

__global__ void k_bn_fin(const float* __restrict__ sums, const float* __restrict__ sumsq,
                         const float* __restrict__ g, const float* __restrict__ bt,
                         float* __restrict__ scale, float* __restrict__ shift,
                         int N, int F) {
    int c = blockIdx.x * blockDim.x + threadIdx.x;
    if (c >= F) return;
    float mu  = sums[c] / (float)N;
    float var = sumsq[c] / (float)N - mu * mu;
    float sc  = g[c] * rsqrtf(var + 1e-5f);
    scale[c] = sc;
    shift[c] = bt[c] - mu * sc;
}

// act = f16(relu(agg*scale + shift))  (next layer's WMMA A operand)
__global__ void k_bn_apply(const float* __restrict__ agg, const float* __restrict__ scale,
                           const float* __restrict__ shift, _Float16* __restrict__ act,
                           int N, int F) {
    int idx = blockIdx.x * blockDim.x + threadIdx.x;
    if (idx >= N * F) return;
    int f = idx % F;
    float v = fmaxf(agg[idx] * scale[f] + shift[f], 0.f);
    act[idx] = (_Float16)v;
}

// ---------------------------------------------------------------- mean pool
__global__ void k_cnt(const int* __restrict__ batch, float* __restrict__ cnt, int N) {
    int i = blockIdx.x * blockDim.x + threadIdx.x;
    if (i < N) atomicAdd(&cnt[batch[i]], 1.f);
}

__global__ void k_pool(const int* __restrict__ batch, const float* __restrict__ h,
                       float* __restrict__ pool, int N) {
    int idx = blockIdx.x * blockDim.x + threadIdx.x;
    if (idx >= N * 32) return;
    int i = idx >> 5, f4 = idx & 31;
    int b = batch[i];
    const float4 v = *(const float4*)(h + (size_t)i * 128 + f4 * 4);
    float* o = pool + (size_t)b * 128 + f4 * 4;
    atomicAdd(o + 0, v.x); atomicAdd(o + 1, v.y);
    atomicAdd(o + 2, v.z); atomicAdd(o + 3, v.w);
}

__global__ void k_enc(const float* __restrict__ pool, const float* __restrict__ cnt,
                      float* __restrict__ enc_out, _Float16* __restrict__ ench, int B) {
    int idx = blockIdx.x * blockDim.x + threadIdx.x;
    if (idx >= B * 128) return;
    int b = idx >> 7;
    float v = pool[idx] / fmaxf(cnt[b], 1.f);
    enc_out[idx] = v;
    ench[idx] = (_Float16)v;
}

// ---------------------------------------------------------------- launch
static inline int cdiv(int a, int b) { return (a + b - 1) / b; }

extern "C" void kernel_launch(void* const* d_in, const int* in_sizes, int n_in,
                              void* d_out, int out_size, void* d_ws, size_t ws_size,
                              hipStream_t stream) {
    const int N = 100000, E = 1600000, B = 1024;

    const float* x     = (const float*)d_in[0];
    const int*   ei    = (const int*)d_in[1];
    const int*   src   = ei;
    const int*   dstI  = ei + E;
    const int*   batch = (const int*)d_in[2];
    const float* W1  = (const float*)d_in[3],  *b1  = (const float*)d_in[4];
    const float* g1  = (const float*)d_in[5],  *bt1 = (const float*)d_in[6];
    const float* W2  = (const float*)d_in[7],  *b2  = (const float*)d_in[8];
    const float* g2  = (const float*)d_in[9],  *bt2 = (const float*)d_in[10];
    const float* W3  = (const float*)d_in[11], *b3  = (const float*)d_in[12];
    const float* Wo1 = (const float*)d_in[13], *bo1 = (const float*)d_in[14];
    const float* Wo2 = (const float*)d_in[15], *bo2 = (const float*)d_in[16];
    const float* Wp1 = (const float*)d_in[17], *bp1 = (const float*)d_in[18];
    const float* Wp2 = (const float*)d_in[19], *bp2 = (const float*)d_in[20];
    const float* Wp3 = (const float*)d_in[21], *bp3 = (const float*)d_in[22];
    const float* Wd1 = (const float*)d_in[23], *bd1 = (const float*)d_in[24];
    const float* Wd2 = (const float*)d_in[25], *bd2 = (const float*)d_in[26];

    float* out = (float*)d_out;
    float* out_op    = out;               // [1024,13]
    float* out_param = out + 13312;       // [1024,1]
    float* out_done  = out + 14336;       // [1024,2]
    float* out_enc   = out + 16384;       // [1024,128]

    // ---- workspace carve (bytes, 256-aligned)
    char* ws = (char*)d_ws;
    size_t o = 0;
    auto carve = [&](size_t bytes) { size_t r = o; o = (o + bytes + 255) & ~(size_t)255; return r; };
    float*    norm  = (float*)(ws + carve((size_t)E * 4));
    float*    dinv  = (float*)(ws + carve((size_t)N * 4));      // deg then dinv
    float*    selfn = (float*)(ws + carve((size_t)N * 4));
    float*    h     = (float*)(ws + carve((size_t)N * 128 * 4));
    float*    agg   = (float*)(ws + carve((size_t)N * 128 * 4));
    _Float16* act   = (_Float16*)(ws + carve((size_t)N * 64 * 2));
    float*    stats = (float*)(ws + carve(512 * 4));            // sums|sumsq|scale|shift
    float*    pool  = (float*)(ws + carve((size_t)B * 128 * 4));
    float*    cnt   = (float*)(ws + carve((size_t)B * 4));
    _Float16* ench  = (_Float16*)(ws + carve((size_t)B * 128 * 2));
    _Float16* hh1   = (_Float16*)(ws + carve((size_t)B * 64 * 2));
    _Float16* hh2   = (_Float16*)(ws + carve((size_t)B * 64 * 2));
    _Float16* wp    = (_Float16*)(ws + carve(65536 * 2));
    (void)ws_size; (void)n_in; (void)in_sizes; (void)out_size;

    float* sums  = stats;       float* sumsq = stats + 128;
    float* scale = stats + 256; float* shift = stats + 384;

    // ---- pack all weights to WMMA B-operand layout (f16)
    struct P { const float* W; int K, Nout, Kt, Nt; size_t off; };
    const P packs[10] = {
        { W1,  20,  64, 1, 4,     0 }, { W2,  64,  64, 2, 4,  2048 },
        { W3,  64, 128, 2, 8,  6144 }, { Wo1,128,  64, 4, 4, 14336 },
        { Wo2, 64,  13, 2, 1, 22528 }, { Wp1,128,  64, 4, 4, 23552 },
        { Wp2, 64,  64, 2, 4, 31744 }, { Wp3, 64,   1, 2, 1, 35840 },
        { Wd1,128,  64, 4, 4, 36864 }, { Wd2, 64,   2, 2, 1, 45056 },
    };
    for (int i = 0; i < 10; ++i) {
        int tot = packs[i].Kt * packs[i].Nt * 512;
        k_pack_w<<<cdiv(tot, 256), 256, 0, stream>>>(packs[i].W, wp + packs[i].off,
                                                     packs[i].K, packs[i].Nout,
                                                     packs[i].Kt, packs[i].Nt);
    }

    // ---- degree / normalization (shared by all three layers)
    k_zero<<<cdiv(N, 256), 256, 0, stream>>>(dinv, N);
    k_deg <<<cdiv(E, 256), 256, 0, stream>>>(dstI, dinv, E);
    k_dinv<<<cdiv(N, 256), 256, 0, stream>>>(dinv, selfn, N);
    k_norm<<<cdiv(E, 256), 256, 0, stream>>>(src, dstI, dinv, norm, E);

    // ---- x -> f16 padded [N,32]
    k_cvt_x<<<cdiv(N * 32, 256), 256, 0, stream>>>(x, act, N);

    const int mtilesN = cdiv(N, 16);
    auto gemm_waves = [&](int M, int Nt) { return cdiv(cdiv(M, 16), MT) * Nt; };

    auto gcn_layer = [&](int lda, size_t woff, int Kt, int Nt, int F,
                         const float* bias, const float* g, const float* bt) {
        (void)mtilesN;
        k_gemm<<<cdiv(gemm_waves(N, Nt), 8), 256, 0, stream>>>(
            act, lda, wp + woff, nullptr, h, F, nullptr, 0, N, Kt, Nt, F, 0);
        k_agg_init<<<cdiv(N * F, 256), 256, 0, stream>>>(h, selfn, bias, agg, N, F);
        int lf4 = (F == 64) ? 4 : 5;
        k_scatter<<<cdiv(E * (F / 4), 256), 256, 0, stream>>>(src, dstI, norm, h, agg, E, lf4);
        if (g) {
            k_zero<<<2, 256, 0, stream>>>(stats, 512);
            k_bn_stats<<<512, 256, 0, stream>>>(agg, sums, sumsq, N, F);
            k_bn_fin<<<1, F, 0, stream>>>(sums, sumsq, g, bt, scale, shift, N, F);
            k_bn_apply<<<cdiv(N * F, 256), 256, 0, stream>>>(agg, scale, shift, act, N, F);
        }
    };

    gcn_layer(32, 0,    1, 4,  64, b1, g1, bt1);   // 20->64, BN+ReLU
    gcn_layer(64, 2048, 2, 4,  64, b2, g2, bt2);   // 64->64, BN+ReLU
    gcn_layer(64, 6144, 2, 8, 128, b3, nullptr, nullptr);  // 64->128

    // ---- global mean pool -> enc (f32 to d_out, f16 for heads)
    k_zero<<<cdiv(B * 128, 256), 256, 0, stream>>>(pool, B * 128);
    k_zero<<<cdiv(B, 256), 256, 0, stream>>>(cnt, B);
    k_cnt <<<cdiv(N, 256), 256, 0, stream>>>(batch, cnt, N);
    k_pool<<<cdiv(N * 32, 256), 256, 0, stream>>>(batch, agg, pool, N);
    k_enc <<<cdiv(B * 128, 256), 256, 0, stream>>>(pool, cnt, out_enc, ench, B);

    // ---- heads (all WMMA; M=1024 -> 64 row tiles, 16 wave-groups)
    // op head
    k_gemm<<<cdiv(gemm_waves(B, 4), 8), 256, 0, stream>>>(ench, 128, wp + 14336, bo1,
        nullptr, 0, hh1, 64, B, 4, 4, 64, 1);
    k_gemm<<<cdiv(gemm_waves(B, 1), 8), 256, 0, stream>>>(hh1, 64, wp + 22528, bo2,
        out_op, 13, nullptr, 0, B, 2, 1, 13, 0);
    // param head
    k_gemm<<<cdiv(gemm_waves(B, 4), 8), 256, 0, stream>>>(ench, 128, wp + 23552, bp1,
        nullptr, 0, hh1, 64, B, 4, 4, 64, 1);
    k_gemm<<<cdiv(gemm_waves(B, 4), 8), 256, 0, stream>>>(hh1, 64, wp + 31744, bp2,
        nullptr, 0, hh2, 64, B, 2, 4, 64, 1);
    k_gemm<<<cdiv(gemm_waves(B, 1), 8), 256, 0, stream>>>(hh2, 64, wp + 35840, bp3,
        out_param, 1, nullptr, 0, B, 2, 1, 1, 0);
    // done head
    k_gemm<<<cdiv(gemm_waves(B, 4), 8), 256, 0, stream>>>(ench, 128, wp + 36864, bd1,
        nullptr, 0, hh1, 64, B, 4, 4, 64, 1);
    k_gemm<<<cdiv(gemm_waves(B, 1), 8), 256, 0, stream>>>(hh1, 64, wp + 45056, bd2,
        out_done, 2, nullptr, 0, B, 2, 1, 2, 0);
}
